// CPFStudent_19765439496457
// MI455X (gfx1250) — matrix operand order repaired and checked by reading, not compiled, not run on hardware
//
#include <hip/hip_runtime.h>
#include <hip/hip_bf16.h>

// ---------------- types for WMMA ----------------
typedef __attribute__((ext_vector_type(16))) __bf16          v16bf;
typedef __attribute__((ext_vector_type(8)))  float           v8f;
typedef __attribute__((ext_vector_type(8)))  unsigned short  v8u;
typedef __attribute__((ext_vector_type(16))) unsigned short  v16u;
typedef __attribute__((ext_vector_type(4)))  unsigned short  v4u;

union FragBF { v16bf bf; v16u us; };

__device__ __forceinline__ unsigned short f2bf(float f) {
    unsigned int u = __float_as_uint(f);
    unsigned int r = (u + 0x7FFFu + ((u >> 16) & 1u)) >> 16;   // RNE
    return (unsigned short)r;
}

// ---------------- constants (match reference) ----------------
#define IN_DIM 512
#define HID    256
#define NC     40
#define NCP    48     // padded to 3 WMMA n-tiles
#define STEPS  10

// ============================================================
// 0) convert weights to bf16 (fc2 padded 40->48 rows with zeros)
// ============================================================
__global__ void conv_w_kernel(const float* __restrict__ w1, const float* __restrict__ w2,
                              unsigned short* __restrict__ o1, unsigned short* __restrict__ o2) {
    int i = blockIdx.x * blockDim.x + threadIdx.x;
    const int n1 = HID * IN_DIM;           // 131072
    const int n2 = NCP * HID;              // 12288
    if (i < n1) {
        o1[i] = f2bf(w1[i]);
    } else if (i < n1 + n2) {
        int j = i - n1;
        int r = j / HID, c = j % HID;
        o2[j] = (r < NC) ? f2bf(w2[r * HID + c]) : (unsigned short)0;
    }
}

// ============================================================
// 1) fused MLP: logits_ft = relu(x@W1^T + b1)@W2^T + b2
//    one wave / block, 16 nodes per block, bf16 WMMA
// ============================================================
__global__ __launch_bounds__(32)
void mlp_wmma_kernel(const float* __restrict__ x,
                     const unsigned short* __restrict__ w1,   // HID x IN_DIM bf16
                     const float* __restrict__ b1,
                     const unsigned short* __restrict__ w2,   // NCP x HID bf16 (padded)
                     const float* __restrict__ b2,
                     float* __restrict__ lft) {
    __shared__ __align__(16) unsigned short xs[16 * IN_DIM];  // 16 KB
    __shared__ __align__(16) unsigned short hs[16 * HID];     // 8 KB
    const int t  = threadIdx.x;            // 0..31
    const int m0 = blockIdx.x * 16;

    // ---- stage x tile (16 x 512 f32) -> bf16 LDS ----
    #pragma unroll
    for (int i = 0; i < 64; ++i) {
        int f  = t + 32 * i;               // float4 id, 2048 total
        int r  = f >> 7;                   // 128 float4 per row
        int cq = f & 127;
        float4 v = *((const float4*)(x + (size_t)(m0 + r) * IN_DIM) + cq);
        v4u u; u[0] = f2bf(v.x); u[1] = f2bf(v.y); u[2] = f2bf(v.z); u[3] = f2bf(v.w);
        *(v4u*)(xs + r * IN_DIM + cq * 4) = u;
    }
    __syncthreads();

    const int lhalf = (t >= 16) ? 1 : 0;
    const int lmod  = t & 15;

    // ---- fc1: 16 n-tiles, K=512 (16 k-steps of 32) ----
    for (int nt = 0; nt < 16; ++nt) {
        v8f acc = {0.f,0.f,0.f,0.f,0.f,0.f,0.f,0.f};
        const unsigned short* wrow = w1 + (size_t)(nt * 16 + lmod) * IN_DIM + lhalf * 8;
        #pragma unroll
        for (int ks = 0; ks < 16; ++ks) {
            int koff = ks * 32 + lhalf * 8;
            v8u a_lo = *(const v8u*)(xs + lmod * IN_DIM + koff);
            v8u a_hi = *(const v8u*)(xs + lmod * IN_DIM + koff + 16);
            v8u b_lo = *(const v8u*)(wrow + ks * 32);
            v8u b_hi = *(const v8u*)(wrow + ks * 32 + 16);
            FragBF a, b;
            a.us = __builtin_shufflevector(a_lo, a_hi, 0,1,2,3,4,5,6,7,8,9,10,11,12,13,14,15);
            b.us = __builtin_shufflevector(b_lo, b_hi, 0,1,2,3,4,5,6,7,8,9,10,11,12,13,14,15);
            acc = __builtin_amdgcn_wmma_f32_16x16x32_bf16(false, a.bf, false, b.bf,
                                                          (short)0, acc, false, false);
        }
        float bias = b1[nt * 16 + lmod];
        #pragma unroll
        for (int j = 0; j < 8; ++j) {
            int M = j + lhalf * 8;
            float h = acc[j] + bias;
            h = h > 0.f ? h : 0.f;         // relu
            hs[M * HID + nt * 16 + lmod] = f2bf(h);
        }
    }
    __syncthreads();

    // ---- fc2: 3 n-tiles (48 cols, 40 valid), K=256 (8 k-steps) ----
    for (int nt = 0; nt < 3; ++nt) {
        v8f acc = {0.f,0.f,0.f,0.f,0.f,0.f,0.f,0.f};
        const unsigned short* wrow = w2 + (size_t)(nt * 16 + lmod) * HID + lhalf * 8;
        #pragma unroll
        for (int ks = 0; ks < 8; ++ks) {
            int koff = ks * 32 + lhalf * 8;
            v8u a_lo = *(const v8u*)(hs + lmod * HID + koff);
            v8u a_hi = *(const v8u*)(hs + lmod * HID + koff + 16);
            v8u b_lo = *(const v8u*)(wrow + ks * 32);
            v8u b_hi = *(const v8u*)(wrow + ks * 32 + 16);
            FragBF a, b;
            a.us = __builtin_shufflevector(a_lo, a_hi, 0,1,2,3,4,5,6,7,8,9,10,11,12,13,14,15);
            b.us = __builtin_shufflevector(b_lo, b_hi, 0,1,2,3,4,5,6,7,8,9,10,11,12,13,14,15);
            acc = __builtin_amdgcn_wmma_f32_16x16x32_bf16(false, a.bf, false, b.bf,
                                                          (short)0, acc, false, false);
        }
        int n = nt * 16 + lmod;
        if (n < NC) {
            float bias = b2[n];
            #pragma unroll
            for (int j = 0; j < 8; ++j) {
                int M = j + lhalf * 8;
                lft[(size_t)(m0 + M) * NC + n] = acc[j] + bias;
            }
        }
    }
}

// ============================================================
// 2) degree / norm precompute
// ============================================================
__global__ void deg_init_kernel(float* __restrict__ dis, int n) {
    int i = blockIdx.x * blockDim.x + threadIdx.x;
    if (i < n) dis[i] = 1.0f;                       // self-loop contributes 1
}
__global__ void deg_acc_kernel(const int* __restrict__ row, float* __restrict__ dis, int e) {
    int i = blockIdx.x * blockDim.x + threadIdx.x;
    if (i < e) atomicAdd(&dis[row[i]], 1.0f);
}
__global__ void deg_rsqrt_kernel(float* __restrict__ dis, int n) {
    int i = blockIdx.x * blockDim.x + threadIdx.x;
    if (i < n) dis[i] = rsqrtf(dis[i]);             // deg >= 1 always (self-loop)
}
__global__ void norm_kernel(const int* __restrict__ row, const int* __restrict__ col,
                            const float* __restrict__ dis, float* __restrict__ nrm, int e) {
    int i = blockIdx.x * blockDim.x + threadIdx.x;
    if (i < e) nrm[i] = dis[row[i]] * dis[col[i]];
}

// ============================================================
// 3) propagation step: zero, scatter (edges + self-loops), mask
// ============================================================
__global__ void zero_kernel(float* __restrict__ dst, int n) {
    int i = blockIdx.x * blockDim.x + threadIdx.x;
    if (i < n) dst[i] = 0.0f;
}
__global__ void scatter_kernel(const int* __restrict__ row, const int* __restrict__ col,
                               const float* __restrict__ nrm, const float* __restrict__ dis,
                               const float* __restrict__ src, float* __restrict__ dst,
                               int e, int n) {
    long long t   = (long long)blockIdx.x * blockDim.x + threadIdx.x;
    long long idx = t >> 3;                         // edge id (8 lanes/edge)
    int lane      = (int)(t & 7);
    if (idx >= (long long)e + n) return;
    int r, c; float w;
    if (idx < e) { r = row[idx]; c = col[idx]; w = nrm[idx]; }
    else { int i = (int)(idx - e); r = c = i; float d = dis[i]; w = d * d; }
    const float* s = src + (size_t)r * NC;
    float*       d = dst + (size_t)c * NC;
    #pragma unroll
    for (int i = 0; i < 5; ++i) {
        int ch = lane + 8 * i;
        atomicAdd(&d[ch], s[ch] * w);
    }
}
__global__ void mask_kernel(const unsigned char* __restrict__ mask,
                            const float* __restrict__ hard, float* __restrict__ dst, int n) {
    int i = blockIdx.x * blockDim.x + threadIdx.x;
    if (i < n && mask[i / NC]) dst[i] = hard[i];
}

// ============================================================
// 4) final blend: out = a*plp + (1-a)*logits_ft, a = sigmoid(alpha)
// ============================================================
__global__ void final_kernel(const float* __restrict__ alpha, const float* __restrict__ plp,
                             const float* __restrict__ lft, float* __restrict__ out, int n) {
    int i = blockIdx.x * blockDim.x + threadIdx.x;
    if (i < n) {
        float a = 1.0f / (1.0f + __expf(-alpha[i / NC]));
        out[i] = a * plp[i] + (1.0f - a) * lft[i];
    }
}

// ============================================================
extern "C" void kernel_launch(void* const* d_in, const int* in_sizes, int n_in,
                              void* d_out, int out_size, void* d_ws, size_t ws_size,
                              hipStream_t stream) {
    const float*         x     = (const float*)d_in[0];
    const int*           eidx  = (const int*)d_in[1];          // (2, E) int32
    const float*         linit = (const float*)d_in[2];
    const unsigned char* tmask = (const unsigned char*)d_in[3];
    const float*         hard  = (const float*)d_in[4];
    const float*         fc1w  = (const float*)d_in[5];
    const float*         fc1b  = (const float*)d_in[6];
    const float*         fc2w  = (const float*)d_in[7];
    const float*         fc2b  = (const float*)d_in[8];
    const float*         alpha = (const float*)d_in[9];
    float* out = (float*)d_out;

    const int N = in_sizes[9];              // alpha has N elements
    const int E = in_sizes[1] / 2;
    const int*  row = eidx;
    const int*  col = eidx + E;
    const long long NCtot = (long long)N * NC;

    // ---- workspace layout (256-B aligned slabs) ----
    char* base = (char*)d_ws;
    size_t off = 0;
    auto take = [&](size_t bytes) -> char* {
        char* p = base + off;
        off = (off + bytes + 255) & ~(size_t)255;
        return p;
    };
    unsigned short* w1b = (unsigned short*)take((size_t)HID * IN_DIM * 2);
    unsigned short* w2b = (unsigned short*)take((size_t)NCP * HID * 2);
    float* dis = (float*)take((size_t)N * 4);
    float* nrm = (float*)take((size_t)E * 4);
    float* lft = (float*)take((size_t)NCtot * 4);
    float* p0  = (float*)take((size_t)NCtot * 4);
    float* p1  = (float*)take((size_t)NCtot * 4);
    (void)ws_size; (void)n_in; (void)out_size;

    const int B = 256;
    // 0) weight conversion
    {
        int tot = HID * IN_DIM + NCP * HID;
        conv_w_kernel<<<(tot + B - 1) / B, B, 0, stream>>>(fc1w, fc2w, w1b, w2b);
    }
    // 1) fused MLP with WMMA (N multiple of 16: 100000 = 6250*16)
    mlp_wmma_kernel<<<N / 16, 32, 0, stream>>>(x, w1b, fc1b, w2b, fc2b, lft);

    // 2) degree + norm
    deg_init_kernel<<<(N + B - 1) / B, B, 0, stream>>>(dis, N);
    deg_acc_kernel<<<(E + B - 1) / B, B, 0, stream>>>(row, dis, E);
    deg_rsqrt_kernel<<<(N + B - 1) / B, B, 0, stream>>>(dis, N);
    norm_kernel<<<(E + B - 1) / B, B, 0, stream>>>(row, col, dis, nrm, E);

    // 3) 10 propagation steps (ping-pong p0/p1; src of step 0 is label_init)
    int ncBlocks = (int)((NCtot + B - 1) / B);
    long long sthreads = ((long long)E + N) * 8;
    int sBlocks = (int)((sthreads + B - 1) / B);
    const float* src = linit;
    float* dst = p0;
    for (int s = 0; s < STEPS; ++s) {
        zero_kernel<<<ncBlocks, B, 0, stream>>>(dst, (int)NCtot);
        scatter_kernel<<<sBlocks, B, 0, stream>>>(row, col, nrm, dis, src, dst, E, N);
        mask_kernel<<<ncBlocks, B, 0, stream>>>(tmask, hard, dst, (int)NCtot);
        src = dst;
        dst = (dst == p0) ? p1 : p0;
    }

    // 4) final blend (src points at the last-written buffer)
    final_kernel<<<ncBlocks, B, 0, stream>>>(alpha, src, lft, out, (int)NCtot);
}
